// Self_Attn_22797686407572
// MI455X (gfx1250) — compile-verified, matrix operand order checked by reading, not run on hardware
//
#include <hip/hip_runtime.h>
#include <hip/hip_bf16.h>

typedef __bf16 bf16;
typedef __attribute__((ext_vector_type(16))) __bf16 v16bf;
typedef __attribute__((ext_vector_type(4)))  __bf16 v4bf;
typedef __attribute__((ext_vector_type(8)))  float  v8f;
typedef unsigned int u32;
typedef unsigned long long u64;

#define BMT 128   // block tile M (4 waves x 32)
#define BNT 64    // block tile N
#define BKT 32    // block tile K (one WMMA k-step)
#define LDBP 20   // dwords per B row (16 k-pairs + 4 pad) -> 80B row, keeps 16B align

// ---------------------------------------------------------------------------
// f32 -> bf16 elementwise conversion, 4 elements/thread
// ---------------------------------------------------------------------------
__global__ __launch_bounds__(256) void cvt_f32_bf16_x4(const float4* __restrict__ in,
                                                       v4bf* __restrict__ out, size_t n4) {
    size_t i = (size_t)blockIdx.x * blockDim.x + threadIdx.x;
    if (i < n4) {
        float4 f = in[i];
        v4bf o;
        o[0] = (bf16)f.x; o[1] = (bf16)f.y; o[2] = (bf16)f.z; o[3] = (bf16)f.w;
        out[i] = o;
    }
}

// ---------------------------------------------------------------------------
// Generic batched bf16 GEMM: Out[b] = A[b](MxK) * B[b](KxN) [+ epilogue]
// MODE 0: bias add, store bf16     (q/k/v projections; A batch-stride 0 = W)
// MODE 1: store f32                (energy, pre-softmax)
// MODE 2: gamma*acc + X, store f32 (final output with residual)
//
// Block: 128 threads = 4 waves stacked along M; block tile 128x64x32.
// Wave tile 32x64 = 2x4 WMMA 16x16x32 tiles (8 WMMA per K-step per wave);
// all waves share the B fragments (broadcast reuse of the B LDS tile).
// A tile staged with GLOBAL_LOAD_ASYNC_TO_LDS_B128 directly into frag-major
// (per-lane WMMA operand order) LDS -> A fragment = one 32B LDS read.
// B tile staged as [n][k-pair] interleaved dwords -> B fragment = 2x ds_load_b128.
// Double-buffered: async A(k+1) + global B(k+1) issued before WMMAs on tile k.
// ---------------------------------------------------------------------------
template <int MODE>
__global__ __launch_bounds__(128) void gemm_bf16_wmma(
    const bf16* __restrict__ A, size_t sAb,
    const bf16* __restrict__ Bm, size_t sBb,
    const float* __restrict__ bias,
    void* __restrict__ Out, size_t sOb,
    const float* __restrict__ X,
    const float* __restrict__ gamma,
    int M, int N, int K)
{
    // A: [buf][sub(8) * 512 + lane*16 + slot*8] bf16, frag-major (32B per lane)
    __shared__ __align__(32) bf16 Asw[2][4096];
    // B: [buf][n(64)][kp dwords], dword = {K=2kp+1 (hi16), K=2kp (lo16)} of col n
    __shared__ __align__(16) u32 Bsp[2][BNT][LDBP];

    const int b  = blockIdx.z;
    const bf16* Ag = A  + (size_t)b * sAb;
    const bf16* Bg = Bm + (size_t)b * sBb;
    const int n0 = blockIdx.x * BNT;
    const int m0 = blockIdx.y * BMT;
    const int tid  = threadIdx.x;
    const int lane = tid & 31;
    const int wave = tid >> 5;
    const int wm = wave * 32;           // wave's M offset within block tile

    // ---- A staging decode: 4 x 16B chunks per thread, frag-major scatter ----
    // chunk c: row = c>>2 (0..127), cpart = c&3 (16B K-chunk within row)
    // dest: sub = row>>4, lane' = (row&15) + (cpart&1)*16, slot = cpart>>1
    const bf16* agbase[4];
    u32 aldsoff[4];
#pragma unroll
    for (int i = 0; i < 4; ++i) {
        int c = tid + i * 128;
        int row = c >> 2, cp = c & 3;
        int sub = row >> 4, mloc = row & 15;
        int ln = mloc + (cp & 1) * 16, sl = cp >> 1;
        agbase[i]  = Ag + (size_t)(m0 + row) * K + cp * 8;
        aldsoff[i] = (u32)(sub * 512 + ln * 16 + sl * 8);   // bf16 elements
    }

    // ---- B staging decode: thread t -> k-pair kp = t>>3, 8 columns n8.. ----
    const int bkp = tid >> 3;
    const int bn8 = (tid & 7) * 8;
    const bf16* bgbase = Bg + (size_t)(2 * bkp) * N + n0 + bn8;
    uint4 blo, bhi;

    auto issueA = [&](int buf, int k0) {
#pragma unroll
        for (int i = 0; i < 4; ++i) {
            u32 lds = (u32)(u64)(&Asw[buf][aldsoff[i]]);
            u64 ga  = (u64)(agbase[i] + k0);
            asm volatile("global_load_async_to_lds_b128 %0, %1, off"
                         :: "v"(lds), "v"(ga) : "memory");
        }
    };
    auto loadB = [&](int k0) {
        const bf16* g = bgbase + (size_t)k0 * N;
        blo = *(const uint4*)g;
        bhi = *(const uint4*)(g + N);
    };
    auto storeB = [&](int buf) {
        u32 L[4] = {blo.x, blo.y, blo.z, blo.w};
        u32 H[4] = {bhi.x, bhi.y, bhi.z, bhi.w};
#pragma unroll
        for (int j = 0; j < 4; ++j) {
            Bsp[buf][bn8 + 2 * j][bkp]     = (L[j] & 0xFFFFu) | (H[j] << 16);
            Bsp[buf][bn8 + 2 * j + 1][bkp] = (L[j] >> 16)     | (H[j] & 0xFFFF0000u);
        }
    };
    auto fragA = [&](int buf, int sub) -> v16bf {
        return *(const v16bf*)&Asw[buf][sub * 512 + lane * 16];
    };
    auto fragB = [&](int buf, int col0) -> v16bf {
        const int half = lane >> 4;
        const u32* row = Bsp[buf][col0 + (lane & 15)];
        union { uint4 q[2]; v16bf f; } u;
        u.q[0] = *(const uint4*)&row[half * 4];       // K 8h .. 8h+7   -> f[0..7]
        u.q[1] = *(const uint4*)&row[8 + half * 4];   // K 16+8h ..+7   -> f[8..15]
        return u.f;
    };

    v8f acc[2][4] = {};
    const int nk = K / BKT;

    // prologue: stage tile 0 into buffer 0
    loadB(0);
    issueA(0, 0);
    storeB(0);
    asm volatile("s_wait_asynccnt 0x0" ::: "memory");
    __syncthreads();

    for (int kt = 0; kt < nk; ++kt) {
        const int cur = kt & 1;
        if (kt + 1 < nk) {            // prefetch next tile into other buffer
            loadB((kt + 1) * BKT);
            issueA(cur ^ 1, (kt + 1) * BKT);
        }

        v16bf a0 = fragA(cur, wave * 2);
        v16bf a1 = fragA(cur, wave * 2 + 1);
        v16bf bf0 = fragB(cur, 0);
        v16bf bf1 = fragB(cur, 16);
        v16bf bf2 = fragB(cur, 32);
        v16bf bf3 = fragB(cur, 48);

        acc[0][0] = __builtin_amdgcn_wmma_f32_16x16x32_bf16(false, a0, false, bf0, (short)0, acc[0][0], false, false);
        acc[0][1] = __builtin_amdgcn_wmma_f32_16x16x32_bf16(false, a0, false, bf1, (short)0, acc[0][1], false, false);
        acc[0][2] = __builtin_amdgcn_wmma_f32_16x16x32_bf16(false, a0, false, bf2, (short)0, acc[0][2], false, false);
        acc[0][3] = __builtin_amdgcn_wmma_f32_16x16x32_bf16(false, a0, false, bf3, (short)0, acc[0][3], false, false);
        acc[1][0] = __builtin_amdgcn_wmma_f32_16x16x32_bf16(false, a1, false, bf0, (short)0, acc[1][0], false, false);
        acc[1][1] = __builtin_amdgcn_wmma_f32_16x16x32_bf16(false, a1, false, bf1, (short)0, acc[1][1], false, false);
        acc[1][2] = __builtin_amdgcn_wmma_f32_16x16x32_bf16(false, a1, false, bf2, (short)0, acc[1][2], false, false);
        acc[1][3] = __builtin_amdgcn_wmma_f32_16x16x32_bf16(false, a1, false, bf3, (short)0, acc[1][3], false, false);

        if (kt + 1 < nk) {
            storeB(cur ^ 1);
            asm volatile("s_wait_asynccnt 0x0" ::: "memory");
        }
        __syncthreads();
    }

    // ---- epilogue: C/D layout: VGPR r -> M = r (+8 for upper half-wave) ----
    const int nOff = lane & 15;
    const int mOff = (lane >> 4) ? 8 : 0;
#pragma unroll
    for (int tm = 0; tm < 2; ++tm) {
#pragma unroll
        for (int tn = 0; tn < 4; ++tn) {
#pragma unroll
            for (int r = 0; r < 8; ++r) {
                int gm = m0 + wm + tm * 16 + mOff + r;
                int gn = n0 + tn * 16 + nOff;
                float val = acc[tm][tn][r];
                size_t idx = (size_t)b * sOb + (size_t)gm * N + gn;
                if (MODE == 0) {
                    val += bias[gm];
                    ((bf16*)Out)[idx] = (bf16)val;
                } else if (MODE == 1) {
                    ((float*)Out)[idx] = val;
                } else {
                    ((float*)Out)[idx] = gamma[0] * val + X[idx];
                }
            }
        }
    }
}

// ---------------------------------------------------------------------------
// Row softmax over energy (f32), writing attn^T as bf16 (fused transpose):
// attnT[b][j][n] = softmax_j(energy[b][n][:])[j]
// One 256-thread block per (b, n) row.
// ---------------------------------------------------------------------------
__global__ __launch_bounds__(256) void softmax_transpose_bf16(
    const float* __restrict__ energy, bf16* __restrict__ attnT, int Nn)
{
    const int row = blockIdx.x;           // row = b*Nn + n
    const int bb = row / Nn;
    const int n  = row - bb * Nn;
    const float* e = energy + ((size_t)bb * Nn + n) * Nn;
    const int j = threadIdx.x;

    __shared__ float red[256];
    float v = e[j];

    red[j] = v;
    __syncthreads();
    for (int s = 128; s > 0; s >>= 1) {
        if (j < s) red[j] = fmaxf(red[j], red[j + s]);
        __syncthreads();
    }
    float mx = red[0];
    __syncthreads();

    float p = __expf(v - mx);
    red[j] = p;
    __syncthreads();
    for (int s = 128; s > 0; s >>= 1) {
        if (j < s) red[j] += red[j + s];
        __syncthreads();
    }
    float denom = red[0];

    attnT[((size_t)bb * Nn + j) * Nn + n] = (bf16)(p / denom);
}

// ---------------------------------------------------------------------------
// Launch
// ---------------------------------------------------------------------------
extern "C" void kernel_launch(void* const* d_in, const int* in_sizes, int n_in,
                              void* d_out, int out_size, void* d_ws, size_t ws_size,
                              hipStream_t stream) {
    const int B = 64, C = 2048, N = 256, Co = 256;  // Co = C/8

    const float* x     = (const float*)d_in[0];
    const float* Wq    = (const float*)d_in[1];
    const float* bq    = (const float*)d_in[2];
    const float* Wk    = (const float*)d_in[3];
    const float* bk    = (const float*)d_in[4];
    const float* Wv    = (const float*)d_in[5];
    const float* bv    = (const float*)d_in[6];
    const float* gamma = (const float*)d_in[7];

    char* ws = (char*)d_ws;
    size_t off = 0;
    auto take = [&](size_t bytes) -> char* {
        char* p = ws + off;
        off = (off + bytes + 255) & ~(size_t)255;
        return p;
    };

    bf16*  xbf    = (bf16*)take((size_t)B * C * N * sizeof(bf16));
    bf16*  Wqb    = (bf16*)take((size_t)Co * C * sizeof(bf16));
    bf16*  Wkb    = (bf16*)take((size_t)Co * C * sizeof(bf16));
    bf16*  Wvb    = (bf16*)take((size_t)C * C * sizeof(bf16));
    bf16*  qb     = (bf16*)take((size_t)B * Co * N * sizeof(bf16));
    bf16*  kb     = (bf16*)take((size_t)B * Co * N * sizeof(bf16));
    bf16*  vb     = (bf16*)take((size_t)B * C * N * sizeof(bf16));
    float* energy = (float*)take((size_t)B * N * N * sizeof(float));
    bf16*  attnT  = (bf16*)take((size_t)B * N * N * sizeof(bf16));

    // 1) cast inputs to bf16 (all sizes divisible by 4)
    {
        size_t n4;
        n4 = (size_t)B * C * N / 4;
        cvt_f32_bf16_x4<<<dim3((n4 + 255) / 256), dim3(256), 0, stream>>>(
            (const float4*)x, (v4bf*)xbf, n4);
        n4 = (size_t)Co * C / 4;
        cvt_f32_bf16_x4<<<dim3((n4 + 255) / 256), dim3(256), 0, stream>>>(
            (const float4*)Wq, (v4bf*)Wqb, n4);
        cvt_f32_bf16_x4<<<dim3((n4 + 255) / 256), dim3(256), 0, stream>>>(
            (const float4*)Wk, (v4bf*)Wkb, n4);
        n4 = (size_t)C * C / 4;
        cvt_f32_bf16_x4<<<dim3((n4 + 255) / 256), dim3(256), 0, stream>>>(
            (const float4*)Wv, (v4bf*)Wvb, n4);
    }

    const size_t sX = (size_t)C * N;    // x / v batch stride
    const size_t sQ = (size_t)Co * N;   // q / k / energy / attnT batch stride

    // 2) projections: q = Wq@x + bq, k = Wk@x + bk  (M=256, K=2048, N=256)
    gemm_bf16_wmma<0><<<dim3(N / BNT, Co / BMT, B), dim3(128), 0, stream>>>(
        Wqb, 0, xbf, sX, bq, qb, sQ, nullptr, nullptr, Co, N, C);
    gemm_bf16_wmma<0><<<dim3(N / BNT, Co / BMT, B), dim3(128), 0, stream>>>(
        Wkb, 0, xbf, sX, bk, kb, sQ, nullptr, nullptr, Co, N, C);
    //    v = Wv@x + bv  (M=2048, K=2048, N=256)
    gemm_bf16_wmma<0><<<dim3(N / BNT, C / BMT, B), dim3(128), 0, stream>>>(
        Wvb, 0, xbf, sX, bv, vb, sX, nullptr, nullptr, C, N, C);

    // 3) energy = q @ k  (M=N=K=256), f32 out
    gemm_bf16_wmma<1><<<dim3(N / BNT, N / BMT, B), dim3(128), 0, stream>>>(
        qb, sQ, kb, sQ, nullptr, energy, sQ, nullptr, nullptr, N, N, N);

    // 4) softmax rows -> attn^T (bf16, transpose fused)
    softmax_transpose_bf16<<<dim3(B * N), dim3(256), 0, stream>>>(energy, attnT, N);

    // 5) out = gamma * (v @ attn^T) + x  (M=2048, K=256, N=256), f32 out
    gemm_bf16_wmma<2><<<dim3(N / BNT, C / BMT, B), dim3(128), 0, stream>>>(
        vb, sX, attnT, sQ, nullptr, d_out, sX, x, gamma, C, N, N);
}
